// BubblePredictor_7799660609595
// MI455X (gfx1250) — matrix operand order repaired and checked by reading, not compiled, not run on hardware
//
#include <hip/hip_runtime.h>
#include <hip/hip_bf16.h>
#include <math.h>

#define B_DIM 1024
#define T_DIM 2048
#define I_DIM 12
#define H_DIM 512

typedef __attribute__((ext_vector_type(16))) __bf16        v16bf;
typedef __attribute__((ext_vector_type(8)))  float         v8f;
typedef __attribute__((ext_vector_type(4)))  unsigned int  v4u;

union Frag {
  v4u   q[2];
  v16bf v;
};

__device__ __forceinline__ unsigned short f2bf(float f) {
  unsigned int u = __float_as_uint(f);
  u += 0x7FFFu + ((u >> 16) & 1u);       // round-to-nearest-even
  return (unsigned short)(u >> 16);
}

__device__ __forceinline__ float sigmoidf_(float x) {
  return 1.0f / (1.0f + __expf(-x));
}

// One-time (per launch) fp32 -> bf16 conversion of w_hh.
__global__ __launch_bounds__(256)
void cvt_bf16_kernel(const float* __restrict__ w, unsigned short* __restrict__ o, int n) {
  int i = blockIdx.x * blockDim.x + threadIdx.x;
  if (i < n) o[i] = f2bf(w[i]);
}

// One GRU time step + (fused) head of the PREVIOUS step.
// Grid: (H/32, B/64); block: 256 threads = 8 wave32s, wave (wr 0..3, wc 0..1)
// owns the 16x16 C tile at rows rowBase+16*wr, cols colBase+16*wc, for all 3 gates.
// Blocks with blockIdx.x == 0 additionally compute logits[., t-1, :] from hcur32
// (the h produced by step t-1) — deterministic shuffle reduction, no atomics.
__global__ __launch_bounds__(256)
void gru_step_kernel(const float* __restrict__ history,        // [B, T, I] fp32
                     const float* __restrict__ w_ih,           // [3H, I]   fp32
                     const unsigned short* __restrict__ whh16, // [3H, H]   bf16
                     const unsigned short* __restrict__ hcur16,// [B, H]    bf16
                     const float* __restrict__ hcur32,         // [B, H]    fp32
                     float* __restrict__ hnew32,               // [B, H]    fp32
                     unsigned short* __restrict__ hnew16,      // [B, H]    bf16
                     const float* __restrict__ w_out,          // [2, H]    fp32
                     const float* __restrict__ b_out,          // [2]       fp32
                     float* __restrict__ logits,               // [B, T, 2] fp32
                     int t) {
  __shared__ float xls[64][I_DIM];        // x tile for this block's 64 rows
  __shared__ float wls[3][32][I_DIM];     // w_ih columns for this block's 32 cols

  const int tid     = threadIdx.x;
  const int rowBase = blockIdx.y * 64;
  const int colBase = blockIdx.x * 32;

  // Stage x tile: history[rowBase+r, t, i]
  for (int e = tid; e < 64 * I_DIM; e += 256) {
    int r = e / I_DIM, i = e % I_DIM;
    xls[r][i] = history[(size_t)(rowBase + r) * (T_DIM * I_DIM) + (size_t)t * I_DIM + i];
  }
  // Stage w_ih tile: w_ih[g*H + colBase + n, i]
  for (int e = tid; e < 3 * 32 * I_DIM; e += 256) {
    int gte = e / (32 * I_DIM);
    int rem = e % (32 * I_DIM);
    int n = rem / I_DIM, i = rem % I_DIM;
    wls[gte][n][i] = w_ih[(size_t)(gte * H_DIM + colBase + n) * I_DIM + i];
  }
  __syncthreads();

  const int wave = tid >> 5;
  const int lane = tid & 31;
  const int wr   = wave >> 1;           // 0..3  (row sub-tile)
  const int wc   = wave & 1;            // 0..1  (col sub-tile)
  const int nl   = lane & 15;           // M for A-frag, N for B/C
  const int g    = lane >> 4;           // K-half selector

  const int aRow = rowBase + wr * 16 + nl;   // A fragment row (M = lane&15)
  const int col  = colBase + wc * 16 + nl;   // B/C column (N = lane&15)

  v8f accR = {}; v8f accZ = {}; v8f accN = {};

  const unsigned short* aPtr = hcur16 + (size_t)aRow * H_DIM;
  const unsigned short* bRp  = whh16 + (size_t)(0 * H_DIM + col) * H_DIM;
  const unsigned short* bZp  = whh16 + (size_t)(1 * H_DIM + col) * H_DIM;
  const unsigned short* bNp  = whh16 + (size_t)(2 * H_DIM + col) * H_DIM;

  // gh = h @ w_hh.T over K = 512, 32 per WMMA.
  // A 16x32 bf16 layout: lane holds M=nl, K = {kb+16g .. kb+16g+15}     -> two b128s.
  // B 32x16 bf16 layout: lane holds N=nl, K = {kb+16g .. kb+16g+15}     -> two b128s.
  #pragma unroll 4
  for (int kb = 0; kb < H_DIM; kb += 32) {
    Frag a, fr, fz, fn;
    a.q[0] = *(const v4u*)(aPtr + kb + 16 * g);
    a.q[1] = *(const v4u*)(aPtr + kb + 16 * g + 8);
    const unsigned short* pr = bRp + kb + 16 * g;
    const unsigned short* pz = bZp + kb + 16 * g;
    const unsigned short* pn = bNp + kb + 16 * g;
    fr.q[0] = *(const v4u*)(pr);     fr.q[1] = *(const v4u*)(pr + 8);
    fz.q[0] = *(const v4u*)(pz);     fz.q[1] = *(const v4u*)(pz + 8);
    fn.q[0] = *(const v4u*)(pn);     fn.q[1] = *(const v4u*)(pn + 8);
    accR = __builtin_amdgcn_wmma_f32_16x16x32_bf16(false, a.v, false, fr.v,
                                                   (short)0, accR, false, false);
    accZ = __builtin_amdgcn_wmma_f32_16x16x32_bf16(false, a.v, false, fz.v,
                                                   (short)0, accZ, false, false);
    accN = __builtin_amdgcn_wmma_f32_16x16x32_bf16(false, a.v, false, fn.v,
                                                   (short)0, accN, false, false);
  }

  // C 16x16 f32 layout: VGPR r, lane -> (M = r + 8*(lane>>4), N = lane&15).
  const int colLoc = wc * 16 + nl;
  #pragma unroll
  for (int r = 0; r < 8; ++r) {
    int mLoc = wr * 16 + r + 8 * g;
    int b    = rowBase + mLoc;
    // Input-gate contributions (exact fp32, K = 12)
    float gir = 0.f, giz = 0.f, gin = 0.f;
    #pragma unroll
    for (int i = 0; i < I_DIM; ++i) {
      float xv = xls[mLoc][i];
      gir = fmaf(xv, wls[0][colLoc][i], gir);
      giz = fmaf(xv, wls[1][colLoc][i], giz);
      gin = fmaf(xv, wls[2][colLoc][i], gin);
    }
    float rg = sigmoidf_(gir + accR[r]);
    float zg = sigmoidf_(giz + accZ[r]);
    float ng = tanhf(gin + rg * accN[r]);          // r gates only the hidden part
    float ho = hcur32[(size_t)b * H_DIM + col];
    float hn = (1.0f - zg) * ng + zg * ho;
    hnew32[(size_t)b * H_DIM + col] = hn;
    hnew16[(size_t)b * H_DIM + col] = f2bf(hn);
  }

  // Fused head for the PREVIOUS step (t-1): logits[b, t-1, :] from hcur32.
  // Uniform branch per block; WMMAs above already executed with full EXEC.
  if (blockIdx.x == 0 && t > 0) {
    const float w00 = 0.0f;  (void)w00;
    #pragma unroll
    for (int rr = 0; rr < 8; ++rr) {
      int b = rowBase + wave * 8 + rr;
      float s0 = 0.f, s1 = 0.f;
      #pragma unroll
      for (int k = lane; k < H_DIM; k += 32) {
        float hv = hcur32[(size_t)b * H_DIM + k];
        s0 = fmaf(hv, w_out[k], s0);
        s1 = fmaf(hv, w_out[H_DIM + k], s1);
      }
      #pragma unroll
      for (int o = 16; o > 0; o >>= 1) {
        s0 += __shfl_xor(s0, o, 32);
        s1 += __shfl_xor(s1, o, 32);
      }
      if (lane == 0) {
        size_t o = (size_t)b * T_DIM * 2 + (size_t)(t - 1) * 2;
        logits[o + 0] = s0 + b_out[0];
        logits[o + 1] = s1 + b_out[1];
      }
    }
  }
}

// Tail: head for the final step (t = T-1) + copy h_final into d_out.
// Grid: B/8 = 128 blocks of 256 threads.
__global__ __launch_bounds__(256)
void gru_tail_kernel(const float* __restrict__ h,       // final h [B, H]
                     const float* __restrict__ w_out,
                     const float* __restrict__ b_out,
                     float* __restrict__ out) {        // [B*T*2 logits | B*H h_final]
  int lane = threadIdx.x & 31;
  int b    = blockIdx.x * 8 + (threadIdx.x >> 5);
  float s0 = 0.f, s1 = 0.f;
  #pragma unroll
  for (int k = lane; k < H_DIM; k += 32) {
    float hv = h[(size_t)b * H_DIM + k];
    s0 = fmaf(hv, w_out[k], s0);
    s1 = fmaf(hv, w_out[H_DIM + k], s1);
  }
  #pragma unroll
  for (int o = 16; o > 0; o >>= 1) {
    s0 += __shfl_xor(s0, o, 32);
    s1 += __shfl_xor(s1, o, 32);
  }
  if (lane == 0) {
    size_t o = (size_t)b * T_DIM * 2 + (size_t)(T_DIM - 1) * 2;
    out[o + 0] = s0 + b_out[0];
    out[o + 1] = s1 + b_out[1];
  }
  // Copy h_final: total B*H floats, 128 blocks * 256 threads -> 16 each.
  float* hf = out + (size_t)B_DIM * T_DIM * 2;
  int base = (blockIdx.x * 256 + threadIdx.x) * 16;
  #pragma unroll
  for (int i = 0; i < 16; ++i) hf[base + i] = h[base + i];
}

extern "C" void kernel_launch(void* const* d_in, const int* in_sizes, int n_in,
                              void* d_out, int out_size, void* d_ws, size_t ws_size,
                              hipStream_t stream) {
  const float* history = (const float*)d_in[0];
  const float* w_ih    = (const float*)d_in[1];
  const float* w_hh    = (const float*)d_in[2];
  const float* w_out   = (const float*)d_in[3];
  const float* b_out   = (const float*)d_in[4];
  float* out = (float*)d_out;

  // Workspace layout (all 256B aligned):
  char* ws = (char*)d_ws;
  size_t off = 0;
  auto take = [&](size_t bytes) -> void* {
    void* p = ws + off;
    off = (off + bytes + 255) & ~(size_t)255;
    return p;
  };
  float*          h32[2];
  unsigned short* h16[2];
  h32[0] = (float*)take((size_t)B_DIM * H_DIM * sizeof(float));
  h32[1] = (float*)take((size_t)B_DIM * H_DIM * sizeof(float));
  h16[0] = (unsigned short*)take((size_t)B_DIM * H_DIM * sizeof(unsigned short));
  h16[1] = (unsigned short*)take((size_t)B_DIM * H_DIM * sizeof(unsigned short));
  unsigned short* whh16 = (unsigned short*)take((size_t)3 * H_DIM * H_DIM * sizeof(unsigned short));

  // h0 = 0 (bf16 zero bits == 0.0f)
  hipMemsetAsync(h32[0], 0, (size_t)B_DIM * H_DIM * sizeof(float), stream);
  hipMemsetAsync(h16[0], 0, (size_t)B_DIM * H_DIM * sizeof(unsigned short), stream);

  // bf16 copy of w_hh (redone every launch: deterministic, cheap, capture-safe)
  int nW = 3 * H_DIM * H_DIM;
  cvt_bf16_kernel<<<(nW + 255) / 256, 256, 0, stream>>>(w_hh, whh16, nW);

  dim3 grid(H_DIM / 32, B_DIM / 64);   // (16, 16) blocks of 8 wave32s
  for (int t = 0; t < T_DIM; ++t) {
    int cur = t & 1;
    int nxt = cur ^ 1;
    gru_step_kernel<<<grid, 256, 0, stream>>>(history, w_ih, whh16,
                                              h16[cur], h32[cur],
                                              h32[nxt], h16[nxt],
                                              w_out, b_out, out, t);
  }

  // T is even: last step (t = T-1, cur = 1) wrote h into buffer 0.
  gru_tail_kernel<<<B_DIM / 8, 256, 0, stream>>>(h32[0], w_out, b_out, out);
}
